// Grapher_63548336112219
// MI455X (gfx1250) — matrix operand order repaired and checked by reading, not compiled, not run on hardware
//
#include <hip/hip_runtime.h>

// ---------------------------------------------------------------------------
// Grapher block (Vision GNN) for gfx1250, wave32, v_wmma_f32_16x16x32_bf16.
// B=32, C=192, C2=384, H=W=28 (N=784), K=9.
// Weights pre-converted to bf16 once (k0) so hot-loop B fragments are single
// 32B global_load_b128s with no VALU conversion.
// ---------------------------------------------------------------------------

typedef __attribute__((ext_vector_type(16))) __bf16 v16bf;
typedef __attribute__((ext_vector_type(8)))  __bf16 v8bf;
typedef __attribute__((ext_vector_type(8)))  float  v8f;

#define B_   32
#define C_   192
#define C2_  384
#define N_   784
#define K_   9
#define EPS_ 1e-5f
#define NT_  (N_/16)          // 49 node tiles per batch

// ---- WMMA fragment helpers (wave32; A/B use contiguous k, not the ISA table order) ----------
// A (16x32 bf16): lane l -> row m=l&15, half h=l>>4; elem j -> k = h*16+j (contiguous)
// B (32x16 bf16): lane l -> col n=l&15, half h=l>>4; elem j -> k = h*16+j (contiguous)
// D (16x16 f32) : lane l -> col n=l&15; elem r -> row m = r + h*8

__device__ __forceinline__ v8f wmma_bf16(v16bf a, v16bf b, v8f c) {
  return __builtin_amdgcn_wmma_f32_16x16x32_bf16(false, a, false, b,
                                                 (short)0, c, false, false);
}

// A fragment from a row-major bf16 row pointer (two contiguous 8-elem runs)
__device__ __forceinline__ v16bf afrag_row_bf16(const __bf16* rp, int kbase, int h) {
  v8bf lo = *(const v8bf*)(rp + kbase + h * 16);
  v8bf hi = *(const v8bf*)(rp + kbase + h * 16 + 8);
  v16bf a;
#pragma unroll
  for (int j = 0; j < 8; ++j) { a[j] = lo[j]; a[j + 8] = hi[j]; }
  return a;
}

// B fragment from 16 contiguous bf16 (single 32B load)
__device__ __forceinline__ v16bf bfrag_bf16(const __bf16* p) {
  return *(const v16bf*)p;
}

// ===========================================================================
// K0: one-shot weight conversion fp32 -> bf16 (removes per-block cvt work)
// ===========================================================================
__global__ void k0_cvt_weights(const float* __restrict__ w1,
                               const float* __restrict__ gw,
                               const float* __restrict__ w2,
                               __bf16* __restrict__ w1b,
                               __bf16* __restrict__ gwb,
                               __bf16* __restrict__ w2b) {
  int i = blockIdx.x * blockDim.x + threadIdx.x;
  if (i < C_ * C_)   w1b[i] = (__bf16)w1[i];
  if (i < C2_ * C2_) gwb[i] = (__bf16)gw[i];
  if (i < C_ * C2_)  w2b[i] = (__bf16)w2[i];
}

// ===========================================================================
// K1: f = BN1(fc1(x));  fn = f / max(||f||,1e-12)  (per node-row of length C)
// x: (B,C,N) f32 ; f: (B,N,C) f32 ; fn: (B,N,C) bf16
// 128 threads (4 waves), each wave owns 3 o-tiles of 16.
// ===========================================================================
#define FS_STR 193
__global__ void k1_fc1_bn_norm(const float* __restrict__ x,
                               const __bf16* __restrict__ w1b,
                               const float* __restrict__ b1,
                               const float* __restrict__ g1,
                               const float* __restrict__ be1,
                               const float* __restrict__ m1,
                               const float* __restrict__ v1,
                               float* __restrict__ f,
                               __bf16* __restrict__ fn) {
  extern __shared__ float sm1[];
  float* Fs = sm1;                         // [16][FS_STR]
  float* ps = Fs + 16 * FS_STR;            // [16][8]
  float* rn = ps + 16 * 8;                 // [16]

  const int blk = blockIdx.x;
  const int b   = blk / NT_;
  const int n0  = (blk - b * NT_) * 16;
  const int t   = threadIdx.x;
  const int w   = t >> 5;
  const int l   = t & 31;
  const int h   = l >> 4;
  const int lm  = l & 15;
  const float* xb = x + (size_t)b * C_ * N_;

  // A fragments: A[m][k] = x[b][k][n0+m]  (strided gather, coalesced across lanes)
  v16bf afr[6];
#pragma unroll
  for (int ks = 0; ks < 6; ++ks) {
    v16bf a;
#pragma unroll
    for (int j = 0; j < 16; ++j) {
      int k = ks * 32 + h * 16 + j;
      a[j] = (__bf16)xb[(size_t)k * N_ + (n0 + lm)];
    }
    afr[ks] = a;
  }

  for (int oi = 0; oi < 3; ++oi) {
    int ot = w * 3 + oi;
    int o  = ot * 16 + lm;
    v8f acc = {0.f, 0.f, 0.f, 0.f, 0.f, 0.f, 0.f, 0.f};
#pragma unroll
    for (int ks = 0; ks < 6; ++ks) {
      v16bf bm = bfrag_bf16(w1b + (size_t)o * C_ + ks * 32 + h * 16);
      acc = wmma_bf16(afr[ks], bm, acc);
    }
    float inv = g1[o] * rsqrtf(v1[o] + EPS_);
    float sh  = b1[o] * inv + be1[o] - m1[o] * inv;
#pragma unroll
    for (int r = 0; r < 8; ++r)
      Fs[(r + h * 8) * FS_STR + o] = acc[r] * inv + sh;
  }
  __syncthreads();

  { // partial sums of squares: 16 rows x 8 chunks of 24
    int m = t & 15, ch = t >> 4;           // t<128 -> ch<8
    float ss = 0.f;
    int c0 = ch * 24;
    for (int c = c0; c < c0 + 24; ++c) { float v = Fs[m * FS_STR + c]; ss += v * v; }
    ps[m * 8 + ch] = ss;
  }
  __syncthreads();
  if (t < 16) {
    float ss = 0.f;
#pragma unroll
    for (int i = 0; i < 8; ++i) ss += ps[t * 8 + i];
    rn[t] = 1.0f / fmaxf(sqrtf(ss), 1e-12f);
  }
  __syncthreads();

  for (int e = t; e < 16 * C_; e += 128) {
    int m = e / C_, c = e - m * C_;
    float v = Fs[m * FS_STR + c];
    size_t row = (size_t)(b * N_ + n0 + m);
    f [row * C_ + c] = v;
    fn[row * C_ + c] = (__bf16)(v * rn[m]);
  }
}

// ===========================================================================
// K2: gram tile G = fn_tile(16xC) * fn_allT(Cx784) via WMMA; top-9 per row.
// (fn normalized => nearest neighbors == largest dot products)
// 256 threads (8 waves); col-tiles 0..48 strided over waves.
// ===========================================================================
#define G_STR 788
__global__ void k2_knn(const __bf16* __restrict__ fn, int* __restrict__ idx) {
  extern __shared__ float sm2[];
  float* G  = sm2;                          // [16][G_STR]
  float* cv = G + 16 * G_STR;               // [16][72]
  int*   ci = (int*)(cv + 16 * 72);         // [16][72]

  const int blk = blockIdx.x;
  const int b   = blk / NT_;
  const int n0  = (blk - b * NT_) * 16;
  const int t   = threadIdx.x;
  const int w   = t >> 5;
  const int l   = t & 31;
  const int h   = l >> 4;
  const int lm  = l & 15;
  const __bf16* fb = fn + (size_t)b * N_ * C_;

  v16bf afr[6];
#pragma unroll
  for (int ks = 0; ks < 6; ++ks)
    afr[ks] = afrag_row_bf16(fb + (size_t)(n0 + lm) * C_, ks * 32, h);

  for (int ct = w; ct < 49; ct += 8) {      // wave-uniform trip count
    int col = ct * 16 + lm;                 // candidate node index (B col)
    v8f acc = {0.f, 0.f, 0.f, 0.f, 0.f, 0.f, 0.f, 0.f};
#pragma unroll
    for (int ks = 0; ks < 6; ++ks) {
      v16bf bm = bfrag_bf16(fb + (size_t)col * C_ + ks * 32 + h * 16);
      acc = wmma_bf16(afr[ks], bm, acc);
    }
#pragma unroll
    for (int r = 0; r < 8; ++r) G[(r + h * 8) * G_STR + col] = acc[r];
  }
  __syncthreads();

  if (t < 128) {                            // 16 rows x 8 chunks of 98 cols
    int m = t & 15, ch = t >> 4;
    float bv[K_]; int bi[K_];
#pragma unroll
    for (int i = 0; i < K_; ++i) { bv[i] = -1e30f; bi[i] = 0; }
    int c0 = ch * 98;
    for (int c = c0; c < c0 + 98; ++c) {
      float vv = G[m * G_STR + c];
      if (vv > bv[K_ - 1]) {
        int p = K_ - 1;
        while (p > 0 && bv[p - 1] < vv) { bv[p] = bv[p - 1]; bi[p] = bi[p - 1]; --p; }
        bv[p] = vv; bi[p] = c;
      }
    }
#pragma unroll
    for (int i = 0; i < K_; ++i) {
      cv[m * 72 + ch * K_ + i] = bv[i];
      ci[m * 72 + ch * K_ + i] = bi[i];
    }
  }
  __syncthreads();

  if (t < 16) {                             // merge 72 candidates -> top-9
    float bv[K_]; int bi[K_];
#pragma unroll
    for (int i = 0; i < K_; ++i) { bv[i] = -1e30f; bi[i] = 0; }
    for (int e = 0; e < 72; ++e) {
      float vv = cv[t * 72 + e]; int ii = ci[t * 72 + e];
      if (vv > bv[K_ - 1]) {
        int p = K_ - 1;
        while (p > 0 && bv[p - 1] < vv) { bv[p] = bv[p - 1]; bi[p] = bi[p - 1]; --p; }
        bv[p] = vv; bi[p] = ii;
      }
    }
#pragma unroll
    for (int i = 0; i < K_; ++i)
      idx[(size_t)(b * N_ + n0 + t) * K_ + i] = bi[i];
  }
}

// ===========================================================================
// K3: edge conv. 16 nodes/block => 144 edges == 9 exact M-tiles of 16.
// e = [x_i, x_j - x_i] (2C) staged bf16 in LDS; per-wave o-tile GEMM;
// BN+ReLU folded at writeback; max over K=9; agg stored bf16 (B,N,C2).
// 256 threads (8 waves), each wave owns 3 of 24 o-tiles.
// ===========================================================================
#define ES_STR 392                          // bf16 units; row = 784B (16B aligned)
#define EH_STR 17
__global__ void k3_edgeconv(const float* __restrict__ f,
                            const int*   __restrict__ idx,
                            const __bf16* __restrict__ gwb,
                            const float* __restrict__ gb,
                            const float* __restrict__ gg,
                            const float* __restrict__ gbe,
                            const float* __restrict__ gm,
                            const float* __restrict__ gv,
                            __bf16* __restrict__ agg) {
  extern __shared__ char sm3raw[];
  __bf16* Es  = (__bf16*)sm3raw;                         // [144][ES_STR]
  float*  ehs = (float*)(sm3raw + 144 * ES_STR * 2);     // [8][144][EH_STR]

  const int blk = blockIdx.x;
  const int b   = blk / NT_;
  const int n0  = (blk - b * NT_) * 16;
  const int t   = threadIdx.x;
  const int w   = t >> 5;
  const int l   = t & 31;
  const int h   = l >> 4;
  const int lm  = l & 15;

  // Build 144 x 384 edge-feature matrix in LDS (bf16)
  for (int e = t; e < 144 * C2_; e += 256) {
    int m = e / C2_, c = e - m * C2_;
    int nl = m / K_, ke = m - nl * K_;
    int n  = n0 + nl;
    const float* fi = f + (size_t)(b * N_ + n) * C_;
    float val;
    if (c < C_) {
      val = fi[c];
    } else {
      int jn = idx[(size_t)(b * N_ + n) * K_ + ke];
      const float* fj = f + (size_t)(b * N_ + jn) * C_;
      val = fj[c - C_] - fi[c - C_];
    }
    Es[m * ES_STR + c] = (__bf16)val;
  }
  __syncthreads();

  for (int oi = 0; oi < 3; ++oi) {
    int ot = w * 3 + oi;
    int o  = ot * 16 + lm;
    float inv = gg[o] * rsqrtf(gv[o] + EPS_);
    float sh  = gb[o] * inv + gbe[o] - gm[o] * inv;

    v16bf bfr[12];
#pragma unroll
    for (int ks = 0; ks < 12; ++ks)
      bfr[ks] = bfrag_bf16(gwb + (size_t)o * C2_ + ks * 32 + h * 16);

    for (int mt = 0; mt < 9; ++mt) {
      v8f acc = {0.f, 0.f, 0.f, 0.f, 0.f, 0.f, 0.f, 0.f};
#pragma unroll
      for (int ks = 0; ks < 12; ++ks) {
        v16bf a = afrag_row_bf16(Es + (size_t)(mt * 16 + lm) * ES_STR, ks * 32, h);
        acc = wmma_bf16(a, bfr[ks], acc);
      }
#pragma unroll
      for (int r = 0; r < 8; ++r) {
        int m = mt * 16 + r + h * 8;      // edge index 0..143
        ehs[((size_t)w * 144 + m) * EH_STR + lm] = fmaxf(acc[r] * inv + sh, 0.0f);
      }
    }
    __syncthreads();
    // max over K per (node, out): lane handles nn=lm, oo=h*8+p
#pragma unroll
    for (int p = 0; p < 8; ++p) {
      int oo = h * 8 + p;
      const float* base = ehs + ((size_t)w * 144 + lm * K_) * EH_STR + oo;
      float mx = base[0];
#pragma unroll
      for (int k = 1; k < K_; ++k) mx = fmaxf(mx, base[(size_t)k * EH_STR]);
      agg[(size_t)(b * N_ + n0 + lm) * C2_ + ot * 16 + oo] = (__bf16)mx;
    }
    __syncthreads();
  }
}

// ===========================================================================
// K4: out = BN2(fc2(agg)) + x   (NCHW output)
// 128 threads (4 waves), 12 o-tiles; A from bf16 agg rows directly.
// ===========================================================================
__global__ void k4_fc2_bn_res(const __bf16* __restrict__ agg,
                              const __bf16* __restrict__ w2b,
                              const float* __restrict__ b2,
                              const float* __restrict__ g2,
                              const float* __restrict__ be2,
                              const float* __restrict__ m2,
                              const float* __restrict__ v2,
                              const float* __restrict__ x,
                              float* __restrict__ out) {
  const int blk = blockIdx.x;
  const int b   = blk / NT_;
  const int n0  = (blk - b * NT_) * 16;
  const int t   = threadIdx.x;
  const int w   = t >> 5;
  const int l   = t & 31;
  const int h   = l >> 4;
  const int lm  = l & 15;

  const __bf16* ar = agg + (size_t)(b * N_ + n0 + lm) * C2_;
  v16bf afr[12];
#pragma unroll
  for (int ks = 0; ks < 12; ++ks) afr[ks] = afrag_row_bf16(ar, ks * 32, h);

  for (int oi = 0; oi < 3; ++oi) {
    int ot = w * 3 + oi;
    int o  = ot * 16 + lm;
    v8f acc = {0.f, 0.f, 0.f, 0.f, 0.f, 0.f, 0.f, 0.f};
#pragma unroll
    for (int ks = 0; ks < 12; ++ks) {
      v16bf bm = bfrag_bf16(w2b + (size_t)o * C2_ + ks * 32 + h * 16);
      acc = wmma_bf16(afr[ks], bm, acc);
    }
    float inv = g2[o] * rsqrtf(v2[o] + EPS_);
    float sh  = b2[o] * inv + be2[o] - m2[o] * inv;
#pragma unroll
    for (int r = 0; r < 8; ++r) {
      int n = n0 + r + h * 8;
      size_t off = ((size_t)b * C_ + o) * N_ + n;
      out[off] = acc[r] * inv + sh + x[off];
    }
  }
}

// ===========================================================================
// Host-side launch
// ===========================================================================
static inline size_t align256(size_t v) { return (v + 255) & ~(size_t)255; }

extern "C" void kernel_launch(void* const* d_in, const int* in_sizes, int n_in,
                              void* d_out, int out_size, void* d_ws, size_t ws_size,
                              hipStream_t stream) {
  const float* x     = (const float*)d_in[0];
  const float* fc1_w = (const float*)d_in[1];
  const float* fc1_b = (const float*)d_in[2];
  const float* bn1_g = (const float*)d_in[3];
  const float* bn1_b = (const float*)d_in[4];
  const float* bn1_m = (const float*)d_in[5];
  const float* bn1_v = (const float*)d_in[6];
  const float* gc_w  = (const float*)d_in[7];
  const float* gc_b  = (const float*)d_in[8];
  const float* gbn_g = (const float*)d_in[9];
  const float* gbn_b = (const float*)d_in[10];
  const float* gbn_m = (const float*)d_in[11];
  const float* gbn_v = (const float*)d_in[12];
  const float* fc2_w = (const float*)d_in[13];
  const float* fc2_b = (const float*)d_in[14];
  const float* bn2_g = (const float*)d_in[15];
  const float* bn2_b = (const float*)d_in[16];
  const float* bn2_m = (const float*)d_in[17];
  const float* bn2_v = (const float*)d_in[18];
  float* out = (float*)d_out;

  // Workspace carve-up
  char* ws = (char*)d_ws;
  float*  f   = (float*)ws;  ws += align256((size_t)B_ * N_ * C_  * sizeof(float));
  __bf16* fn  = (__bf16*)ws; ws += align256((size_t)B_ * N_ * C_  * sizeof(__bf16));
  int*    idx = (int*)ws;    ws += align256((size_t)B_ * N_ * K_  * sizeof(int));
  __bf16* agg = (__bf16*)ws; ws += align256((size_t)B_ * N_ * C2_ * sizeof(__bf16));
  __bf16* w1b = (__bf16*)ws; ws += align256((size_t)C_ * C_   * sizeof(__bf16));
  __bf16* gwb = (__bf16*)ws; ws += align256((size_t)C2_ * C2_ * sizeof(__bf16));
  __bf16* w2b = (__bf16*)ws; ws += align256((size_t)C_ * C2_  * sizeof(__bf16));

  const int grid = B_ * NT_;                           // 1568 blocks

  // K0: weight conversion (covers the largest weight matrix, C2*C2)
  k0_cvt_weights<<<(C2_ * C2_ + 255) / 256, 256, 0, stream>>>(
      fc1_w, gc_w, fc2_w, w1b, gwb, w2b);

  size_t sh1 = (size_t)(16 * FS_STR + 16 * 8 + 16) * sizeof(float);
  k1_fc1_bn_norm<<<grid, 128, sh1, stream>>>(x, w1b, fc1_b, bn1_g, bn1_b,
                                             bn1_m, bn1_v, f, fn);

  size_t sh2 = (size_t)(16 * G_STR) * sizeof(float)
             + (size_t)(16 * 72) * sizeof(float)
             + (size_t)(16 * 72) * sizeof(int);
  k2_knn<<<grid, 256, sh2, stream>>>(fn, idx);

  size_t sh3 = (size_t)(144 * ES_STR) * sizeof(__bf16)
             + (size_t)(8 * 144 * EH_STR) * sizeof(float);
  k3_edgeconv<<<grid, 256, sh3, stream>>>(f, idx, gwb, gc_b, gbn_g, gbn_b,
                                          gbn_m, gbn_v, agg);

  k4_fc2_bn_res<<<grid, 128, 0, stream>>>(agg, w2b, fc2_b, bn2_g, bn2_b,
                                          bn2_m, bn2_v, x, out);
}
